// ConditionalCrossAttentionBlock_53154515255724
// MI455X (gfx1250) — matrix operand order repaired and checked by reading, not compiled, not run on hardware
//
#include <hip/hip_runtime.h>
#include <hip/hip_bf16.h>

#define DIMX   2048
#define KVD    1024
#define NH     16
#define DH     128
#define BATCH  2
#define LQ_    4096
#define LKV_   1024
#define EPSF   1e-6f

typedef __attribute__((ext_vector_type(16))) _Float16 v16h;
typedef __attribute__((ext_vector_type(8)))  _Float16 v8h;
typedef __attribute__((ext_vector_type(4)))  _Float16 v4h;
typedef __attribute__((ext_vector_type(8)))  float    v8f;
typedef __attribute__((ext_vector_type(4)))  unsigned v4u;
typedef __attribute__((ext_vector_type(4)))  int      v4i;
typedef __attribute__((ext_vector_type(8)))  int      v8i;

#if defined(__HIP_DEVICE_COMPILE__) && __has_builtin(__builtin_amdgcn_tensor_load_to_lds)
#define HAVE_TDM 1
#else
#define HAVE_TDM 0
#endif

__device__ __forceinline__ v16h pack16(v8h lo, v8h hi) {
    v16h r;
#pragma unroll
    for (int j = 0; j < 8; ++j) { r[j] = lo[j]; r[8 + j] = hi[j]; }
    return r;
}

__device__ __forceinline__ v8f wmma16(v16h a, v16h b, v8f c) {
    // D = A(16x32 f16) * B(32x16 f16) + C(16x16 f32)
    return __builtin_amdgcn_wmma_f32_16x16x32_f16(
        /*neg_a=*/false, a, /*neg_b=*/false, b,
        /*c_mod=*/(short)0, c, /*reuse_a=*/false, /*reuse_b=*/false);
}

#if HAVE_TDM
// ---------------------------------------------------------------------------
// TDM: DMA a 2D tile (tile_d1 rows x tile_d0 f16 elems, row stride = stride
// elems in global) into LDS, rows packed consecutively. D# packing per
// CDNA5 ISA (group0 §8.3, group1 §8.4). Tracked: TENSORcnt.
// ---------------------------------------------------------------------------
__device__ __forceinline__ void tdm_load_2d_f16(unsigned lds_off, const _Float16* gptr,
                                                unsigned tensor_d0, unsigned tensor_d1,
                                                unsigned tile_d0, unsigned tile_d1,
                                                unsigned stride_elems) {
    const unsigned long long ga = (unsigned long long)(size_t)(const void*)gptr;
    v4u g0;
    g0[0] = 1u;                                           // count=1 valid user D#
    g0[1] = lds_off;                                      // lds_addr (bytes)
    g0[2] = (unsigned)(ga & 0xffffffffu);                 // global_addr[31:0]
    g0[3] = (unsigned)((ga >> 32) & 0x01ffffffu)          // global_addr[56:32]
          | 0x80000000u;                                  // type=2 ("image")
    v8i g1;
    g1[0] = (int)(1u << 16);                              // data_size=1 (2B), no mcast
    g1[1] = (int)((tensor_d0 & 0xffffu) << 16);           // tensor_dim0[15:0]
    g1[2] = (int)(((tensor_d0 >> 16) & 0xffffu) |         // tensor_dim0[31:16]
                  ((tensor_d1 & 0xffffu) << 16));         // tensor_dim1[15:0]
    g1[3] = (int)(((tensor_d1 >> 16) & 0xffffu) |         // tensor_dim1[31:16]
                  ((tile_d0 & 0xffffu) << 16));           // tile_dim0
    g1[4] = (int)(tile_d1 & 0xffffu);                     // tile_dim1; tile_dim2=0
    g1[5] = (int)stride_elems;                            // tensor_dim0_stride[31:0]
    g1[6] = 0;                                            // stride hi / dim1_stride lo
    g1[7] = 0;
    const v4i gz = {0, 0, 0, 0};
#if defined(__clang_major__) && (__clang_major__ >= 23)
    const v8i gz8 = {0, 0, 0, 0, 0, 0, 0, 0};
    __builtin_amdgcn_tensor_load_to_lds(g0, g1, gz, gz, gz8, 0);
#else
    __builtin_amdgcn_tensor_load_to_lds(g0, g1, gz, gz, 0);
#endif
}
#endif // HAVE_TDM

// ---------------------------------------------------------------------------
// fp32 -> fp16 elementwise convert (vectorized x4)
// ---------------------------------------------------------------------------
__global__ void __launch_bounds__(256)
cvt_f32_f16_kernel(const float* __restrict__ in, _Float16* __restrict__ out, long n4) {
    long i = (long)blockIdx.x * 256 + threadIdx.x;
    if (i >= n4) return;
    const float4 vf = ((const float4*)in)[i];
    v4h o;
    o[0] = (_Float16)vf.x; o[1] = (_Float16)vf.y;
    o[2] = (_Float16)vf.z; o[3] = (_Float16)vf.w;
    ((v4h*)out)[i] = o;
}

// ---------------------------------------------------------------------------
// LayerNorm over KVD=1024 columns, one block per row, output f16
// ---------------------------------------------------------------------------
__global__ void __launch_bounds__(256)
ln_kernel(const float* __restrict__ y, const float* __restrict__ w,
          const float* __restrict__ b, _Float16* __restrict__ out) {
    __shared__ float r1[256], r2[256];
    const int tid = threadIdx.x;
    const size_t base = (size_t)blockIdx.x * KVD;
    const float4 vv = *(const float4*)(y + base + tid * 4);
    float v[4] = { vv.x, vv.y, vv.z, vv.w };
    float s = 0.f, sq = 0.f;
#pragma unroll
    for (int i = 0; i < 4; ++i) { s += v[i]; sq += v[i] * v[i]; }
    r1[tid] = s; r2[tid] = sq;
    __syncthreads();
    for (int st = 128; st > 0; st >>= 1) {
        if (tid < st) { r1[tid] += r1[tid + st]; r2[tid] += r2[tid + st]; }
        __syncthreads();
    }
    const float mu  = r1[0] * (1.0f / KVD);
    const float var = r2[0] * (1.0f / KVD) - mu * mu;
    const float inv = rsqrtf(var + EPSF);
    v4h o;
#pragma unroll
    for (int i = 0; i < 4; ++i) {
        const int e = tid * 4 + i;
        o[i] = (_Float16)((v[i] - mu) * inv * w[e] + b[e]);
    }
    *(v4h*)(out + base + tid * 4) = o;
}

// ---------------------------------------------------------------------------
// Fused RMSNorm (over DIMX) + per-head RoPE, f16 in / f16 out, block per row
// ---------------------------------------------------------------------------
__global__ void __launch_bounds__(256)
rmsrope_kernel(const _Float16* __restrict__ in, const float* __restrict__ w,
               const float* __restrict__ cosb, const float* __restrict__ sinb,
               _Float16* __restrict__ out) {
    __shared__ float red[256];
    const int tid = threadIdx.x;
    const size_t base = (size_t)blockIdx.x * DIMX;
    const int e0 = tid * 8;                 // 8 contiguous elems, same head, same half
    v8h vh = *(const v8h*)(in + base + e0);
    float v[8], ss = 0.f;
#pragma unroll
    for (int i = 0; i < 8; ++i) { v[i] = (float)vh[i]; ss += v[i] * v[i]; }
    red[tid] = ss;
    __syncthreads();
    for (int st = 128; st > 0; st >>= 1) {
        if (tid < st) red[tid] += red[tid + st];
        __syncthreads();
    }
    const float inv = rsqrtf(red[0] * (1.0f / DIMX) + EPSF);
    const int pe0 = e0 ^ 64;                // rotate_half partner chunk
    const v8h ph = *(const v8h*)(in + base + pe0);
    const int d = e0 & (DH - 1);
    const float sgn = (d < 64) ? -1.f : 1.f;
    const float* cp = cosb + (size_t)blockIdx.x * DH + d;  // [B*L, DH] broadcast layout
    const float* sp = sinb + (size_t)blockIdx.x * DH + d;
    v8h o;
#pragma unroll
    for (int i = 0; i < 8; ++i) {
        const float t  = v[i] * inv * w[e0 + i];
        const float pr = (float)ph[i] * inv * w[pe0 + i];
        o[i] = (_Float16)(t * cp[i] + sgn * pr * sp[i]);
    }
    *(v8h*)(out + base + e0) = o;
}

// ---------------------------------------------------------------------------
// C[M,N] = A[M,K] * W[N,K]^T + bias   (f16 in, fp32 accum, OutT out)
// 128x128 block tile, BK=32, 8 waves -> each wave 64x32 subtile (4x2 wmma).
// Staging: TDM double-buffered (tensor_load_to_lds + s_wait_tensorcnt) when
// available; per-lane global loads otherwise.
// ---------------------------------------------------------------------------
template <typename OutT>
__global__ void __launch_bounds__(256)
gemm_wmma(const _Float16* __restrict__ A, const _Float16* __restrict__ W,
          const float* __restrict__ bias, OutT* __restrict__ C,
          int M, int N, int K) {
    constexpr int BK = 32;
    const int tid = threadIdx.x, lane = tid & 31, wave = tid >> 5;
    const int wm = wave >> 2, wn = wave & 3;       // 2 (M) x 4 (N) wave grid
    const int lh = lane >> 4, lm = lane & 15;
    const int bm0 = blockIdx.y * 128, bn0 = blockIdx.x * 128;

    const v8f vzero = {0.f, 0.f, 0.f, 0.f, 0.f, 0.f, 0.f, 0.f};
    v8f acc[4][2];
#pragma unroll
    for (int mt = 0; mt < 4; ++mt)
#pragma unroll
        for (int nt = 0; nt < 2; ++nt) acc[mt][nt] = vzero;

#if HAVE_TDM
    __shared__ __align__(16) _Float16 lA[2][128 * BK];
    __shared__ __align__(16) _Float16 lW[2][128 * BK];
    const bool issuer = (wave == 0);
    if (issuer) {   // prologue: stream first K-tile pair into buffer 0
        tdm_load_2d_f16((unsigned)(size_t)&lA[0][0], A + (size_t)bm0 * K,
                        (unsigned)K, (unsigned)(M - bm0), BK, 128, (unsigned)K);
        tdm_load_2d_f16((unsigned)(size_t)&lW[0][0], W + (size_t)bn0 * K,
                        (unsigned)K, (unsigned)(N - bn0), BK, 128, (unsigned)K);
    }
    int buf = 0;
    for (int kk = 0; kk < K; kk += BK, buf ^= 1) {
        if (issuer) {
            if (kk + BK < K) {      // stream next tile pair into other buffer
                tdm_load_2d_f16((unsigned)(size_t)&lA[buf ^ 1][0],
                                A + (size_t)bm0 * K + kk + BK,
                                (unsigned)K, (unsigned)(M - bm0), BK, 128, (unsigned)K);
                tdm_load_2d_f16((unsigned)(size_t)&lW[buf ^ 1][0],
                                W + (size_t)bn0 * K + kk + BK,
                                (unsigned)K, (unsigned)(N - bn0), BK, 128, (unsigned)K);
                __builtin_amdgcn_s_wait_tensorcnt(2);   // current pair resolved
            } else {
                __builtin_amdgcn_s_wait_tensorcnt(0);
            }
        }
        __syncthreads();
        const _Float16* cA = &lA[buf][0];
        const _Float16* cW = &lW[buf][0];
#else
    __shared__ __align__(16) _Float16 lA[128 * BK];
    __shared__ __align__(16) _Float16 lW[128 * BK];
    const int lrow = tid >> 1, lcol = (tid & 1) * 16;
    const _Float16* gA = A + (size_t)(bm0 + lrow) * K + lcol;
    const _Float16* gW = W + (size_t)(bn0 + lrow) * K + lcol;
    for (int kk = 0; kk < K; kk += BK) {
        *(v8h*)(lA + lrow * BK + lcol)     = *(const v8h*)(gA + kk);
        *(v8h*)(lA + lrow * BK + lcol + 8) = *(const v8h*)(gA + kk + 8);
        *(v8h*)(lW + lrow * BK + lcol)     = *(const v8h*)(gW + kk);
        *(v8h*)(lW + lrow * BK + lcol + 8) = *(const v8h*)(gW + kk + 8);
        if (kk + BK < K) {
            __builtin_prefetch(gA + kk + BK, 0, 0);   // global_prefetch_b8
            __builtin_prefetch(gW + kk + BK, 0, 0);
        }
        __syncthreads();
        const _Float16* cA = lA;
        const _Float16* cW = lW;
#endif
        v16h af[4], bf[2];
#pragma unroll
        for (int mt = 0; mt < 4; ++mt) {              // A frag: lane=M row, K split by lh
            const _Float16* p = cA + (wm * 64 + mt * 16 + lm) * BK;
            af[mt] = pack16(*(const v8h*)(p + lh * 16), *(const v8h*)(p + lh * 16 + 8));
        }
#pragma unroll
        for (int nt = 0; nt < 2; ++nt) {              // B frag: lane=N col, K contiguous
            const _Float16* p = cW + (wn * 32 + nt * 16 + lm) * BK + lh * 16;
            bf[nt] = pack16(*(const v8h*)p, *(const v8h*)(p + 8));
        }
#pragma unroll
        for (int mt = 0; mt < 4; ++mt)
#pragma unroll
            for (int nt = 0; nt < 2; ++nt)
                acc[mt][nt] = wmma16(af[mt], bf[nt], acc[mt][nt]);
        __syncthreads();
    }

    // D layout: lane -> N = lm, M = 8*lh + i
#pragma unroll
    for (int mt = 0; mt < 4; ++mt) {
        const int gm = bm0 + wm * 64 + mt * 16 + lh * 8;
#pragma unroll
        for (int nt = 0; nt < 2; ++nt) {
            const int gn = bn0 + wn * 32 + nt * 16 + lm;
            const float bb = bias[gn];
#pragma unroll
            for (int i = 0; i < 8; ++i)
                C[(size_t)(gm + i) * N + gn] = (OutT)(acc[mt][nt][i] + bb);
        }
    }
}

// ---------------------------------------------------------------------------
// Flash attention: grid (LQ/128, H, B), 8 waves x 16 q-rows each.
// S^T = K*Q^T (kv in M dim -> softmax stats per-lane + one shfl_xor(16)),
// O^T += V^T * P^T (P bounced through wave-private LDS [q][kv] tile).
// ---------------------------------------------------------------------------
__global__ void __launch_bounds__(256)
attn_kernel(const _Float16* __restrict__ q, const _Float16* __restrict__ k,
            const _Float16* __restrict__ v, _Float16* __restrict__ o) {
    __shared__ __align__(16) _Float16 lK[32 * DH];     // [kv][d]
    __shared__ __align__(16) _Float16 lVt[DH * 32];    // [d][kv]
    __shared__ __align__(16) _Float16 lP[8 * 16 * 32]; // per-wave [q][kv]

    const int tid = threadIdx.x, lane = tid & 31, wave = tid >> 5;
    const int lh = lane >> 4, lm = lane & 15;
    const int b = blockIdx.z, h = blockIdx.y;
    const int q0 = blockIdx.x * 128 + wave * 16;

    // Q^T B-fragments (lane = q col, d contiguous), kept resident all kv-loop
    v16h qb[4];
    {
        const _Float16* qp = q + ((size_t)(b * LQ_ + q0 + lm) * NH + h) * DH;
#pragma unroll
        for (int kb = 0; kb < 4; ++kb) {
            const _Float16* pp = qp + kb * 32 + lh * 16;
            qb[kb] = pack16(*(const v8h*)pp, *(const v8h*)(pp + 8));
        }
    }

    const v8f vzero = {0.f, 0.f, 0.f, 0.f, 0.f, 0.f, 0.f, 0.f};
    v8f oacc[8];
#pragma unroll
    for (int t = 0; t < 8; ++t) oacc[t] = vzero;
    float rm = -3.0e38f, rl = 0.f;
    const float scale = 0.08838834764831845f;   // 1/sqrt(128)

    const int sr = tid >> 3;             // K staging: kv row 0..31
    const int sc = (tid & 7) * 16;       // K staging: d chunk
    const int vkv = (tid & 15) * 2;      // V staging: kv row pair
    const int vdc = (tid >> 4) * 8;      // V staging: 8-d chunk

    for (int kv0 = 0; kv0 < LKV_; kv0 += 32) {
        {   // stage K straight
            const _Float16* gk = k + ((size_t)(b * LKV_ + kv0 + sr) * NH + h) * DH + sc;
            *(v8h*)(lK + sr * DH + sc)     = *(const v8h*)gk;
            *(v8h*)(lK + sr * DH + sc + 8) = *(const v8h*)(gk + 8);
            // stage V transposed: two kv rows -> packed 2xf16 b32 stores
            const _Float16* gv0 = v + ((size_t)(b * LKV_ + kv0 + vkv) * NH + h) * DH + vdc;
            const _Float16* gv1 = gv0 + (size_t)NH * DH;
            const v8h a0 = *(const v8h*)gv0, a1 = *(const v8h*)gv1;
            unsigned* lVt32 = (unsigned*)lVt;
#pragma unroll
            for (int j = 0; j < 8; ++j) {
                union { _Float16 h2[2]; unsigned u; } cv;
                cv.h2[0] = a0[j]; cv.h2[1] = a1[j];
                lVt32[(vdc + j) * 16 + (vkv >> 1)] = cv.u;
            }
        }
        __syncthreads();

        // S^T[kv][q], 2 kv-tiles x 4 K-steps over Dh
        v8f s[2]; s[0] = vzero; s[1] = vzero;
#pragma unroll
        for (int t = 0; t < 2; ++t)
#pragma unroll
            for (int kb = 0; kb < 4; ++kb) {
                const _Float16* pp = lK + (t * 16 + lm) * DH + kb * 32;
                v16h a = pack16(*(const v8h*)(pp + lh * 16),
                                *(const v8h*)(pp + lh * 16 + 8));
                s[t] = wmma16(a, qb[kb], s[t]);
            }

        // online softmax; lane holds q=lm, kv half split with partner lane^16
        float mloc = -3.0e38f;
#pragma unroll
        for (int t = 0; t < 2; ++t)
#pragma unroll
            for (int i = 0; i < 8; ++i) {
                const float sv = s[t][i] * scale;
                s[t][i] = sv;
                mloc = fmaxf(mloc, sv);
            }
        mloc = fmaxf(mloc, __shfl_xor(mloc, 16, 32));
        const float mnew  = fmaxf(rm, mloc);
        const float alpha = __expf(rm - mnew);
        float psum = 0.f;
        v8h p0, p1;
#pragma unroll
        for (int i = 0; i < 8; ++i) {
            const float ea = __expf(s[0][i] - mnew);
            const float eb = __expf(s[1][i] - mnew);
            psum += ea + eb;
            p0[i] = (_Float16)ea; p1[i] = (_Float16)eb;
        }
        psum += __shfl_xor(psum, 16, 32);
        rl = rl * alpha + psum;
        rm = mnew;
#pragma unroll
        for (int t = 0; t < 8; ++t)
#pragma unroll
            for (int i = 0; i < 8; ++i) oacc[t][i] *= alpha;

        // P^T through wave-private LDS: store [q=lm][kv], reload as B-frag
        _Float16* pw = lP + wave * 512 + lm * 32;
        *(v8h*)(pw + lh * 8)      = p0;   // kv = 8*lh + i
        *(v8h*)(pw + 16 + lh * 8) = p1;   // kv = 16 + 8*lh + i
        const _Float16* prd = lP + wave * 512 + lm * 32 + lh * 16;
        const v16h pb = pack16(*(const v8h*)prd, *(const v8h*)(prd + 8));

        // O^T[d][q] += V^T * P^T, 8 d-tiles
#pragma unroll
        for (int t = 0; t < 8; ++t) {
            const _Float16* pp = lVt + (t * 16 + lm) * 32;
            v16h a = pack16(*(const v8h*)(pp + lh * 16),
                            *(const v8h*)(pp + lh * 16 + 8));
            oacc[t] = wmma16(a, pb, oacc[t]);
        }
        __syncthreads();
    }

    const float invl = 1.0f / rl;
    _Float16* go = o + ((size_t)(b * LQ_ + q0 + lm) * NH + h) * DH;
#pragma unroll
    for (int t = 0; t < 8; ++t) {
        v8h st;
#pragma unroll
        for (int i = 0; i < 8; ++i) st[i] = (_Float16)(oacc[t][i] * invl);
        *(v8h*)(go + t * 16 + lh * 8) = st;   // d = 16t + 8*lh + i
    }
}

// ---------------------------------------------------------------------------
extern "C" void kernel_launch(void* const* d_in, const int* in_sizes, int n_in,
                              void* d_out, int out_size, void* d_ws, size_t ws_size,
                              hipStream_t stream) {
    (void)in_sizes; (void)n_in; (void)out_size; (void)ws_size;
    const float* x      = (const float*)d_in[0];
    const float* y      = (const float*)d_in[1];
    const float* x_cos  = (const float*)d_in[2];
    const float* x_sin  = (const float*)d_in[3];
    const float* y_cos  = (const float*)d_in[4];
    const float* y_sin  = (const float*)d_in[5];
    const float* Wq     = (const float*)d_in[6];
    const float* bq     = (const float*)d_in[7];
    const float* Wk     = (const float*)d_in[8];
    const float* bk     = (const float*)d_in[9];
    const float* Wv     = (const float*)d_in[10];
    const float* bv     = (const float*)d_in[11];
    const float* Wo     = (const float*)d_in[12];
    const float* bo     = (const float*)d_in[13];
    const float* rmsq_w = (const float*)d_in[14];
    const float* rmsk_w = (const float*)d_in[15];
    const float* ln_w   = (const float*)d_in[16];
    const float* ln_b   = (const float*)d_in[17];
    float* out = (float*)d_out;

    const size_t MQ = (size_t)BATCH * LQ_;    // 8192
    const size_t MK = (size_t)BATCH * LKV_;   // 2048

    _Float16* p = (_Float16*)d_ws;
    _Float16* x_h    = p; p += MQ * DIMX;
    _Float16* Wq_h   = p; p += (size_t)DIMX * DIMX;
    _Float16* Wk_h   = p; p += (size_t)DIMX * KVD;
    _Float16* Wv_h   = p; p += (size_t)DIMX * KVD;
    _Float16* Wo_h   = p; p += (size_t)DIMX * DIMX;
    _Float16* yn_h   = p; p += MK * KVD;
    _Float16* qlin_h = p; p += MQ * DIMX;
    _Float16* klin_h = p; p += MK * DIMX;
    _Float16* v_h    = p; p += MK * DIMX;
    _Float16* q_h    = p; p += MQ * DIMX;
    _Float16* k_h    = p; p += MK * DIMX;
    _Float16* ao_h   = qlin_h;   // alias: q-linear is dead after RoPE

    auto cvt = [&](const float* in, _Float16* o, size_t n) {
        const long n4 = (long)(n / 4);
        cvt_f32_f16_kernel<<<dim3((unsigned)((n4 + 255) / 256)), dim3(256), 0, stream>>>(in, o, n4);
    };
    cvt(x,  x_h,  MQ * DIMX);
    cvt(Wq, Wq_h, (size_t)DIMX * DIMX);
    cvt(Wk, Wk_h, (size_t)DIMX * KVD);
    cvt(Wv, Wv_h, (size_t)DIMX * KVD);
    cvt(Wo, Wo_h, (size_t)DIMX * DIMX);

    ln_kernel<<<dim3((unsigned)MK), dim3(256), 0, stream>>>(y, ln_w, ln_b, yn_h);

    // q_lin = x @ Wq^T + bq ; k_lin = yn @ Wk^T + bk ; v = yn @ Wv^T + bv
    gemm_wmma<_Float16><<<dim3(16, 64), dim3(256), 0, stream>>>(x_h,  Wq_h, bq, qlin_h, (int)MQ, DIMX, DIMX);
    gemm_wmma<_Float16><<<dim3(16, 16), dim3(256), 0, stream>>>(yn_h, Wk_h, bk, klin_h, (int)MK, DIMX, KVD);
    gemm_wmma<_Float16><<<dim3(16, 16), dim3(256), 0, stream>>>(yn_h, Wv_h, bv, v_h,    (int)MK, DIMX, KVD);

    rmsrope_kernel<<<dim3((unsigned)MQ), dim3(256), 0, stream>>>(qlin_h, rmsq_w, x_cos, x_sin, q_h);
    rmsrope_kernel<<<dim3((unsigned)MK), dim3(256), 0, stream>>>(klin_h, rmsk_w, y_cos, y_sin, k_h);

    attn_kernel<<<dim3(LQ_ / 128, NH, BATCH), dim3(256), 0, stream>>>(q_h, k_h, v_h, ao_h);

    // out = attn @ Wo^T + bo  (fp32 to d_out)
    gemm_wmma<float><<<dim3(16, 64), dim3(256), 0, stream>>>(ao_h, Wo_h, bo, out, (int)MQ, DIMX, DIMX);
}
